// CausalSelfAttention_44787918962796
// MI455X (gfx1250) — compile-verified, matrix operand order checked by reading, not run on hardware
//
#include <hip/hip_runtime.h>
#include <hip/hip_bf16.h>

// ---------------------------------------------------------------------------
// CDNA5 (gfx1250) causal sliding-window attention block.
// Matmuls on v_wmma_f32_16x16x32_bf16 (f32 accumulate); GEMM operand staging
// via global_load_async_to_lds_b128 (ASYNCcnt) with double-buffered LDS.
// ---------------------------------------------------------------------------

typedef __attribute__((ext_vector_type(16))) __bf16 v16bf;
typedef __attribute__((ext_vector_type(8)))  __bf16 v8bf;
typedef __attribute__((ext_vector_type(8)))  float  v8f;

union BFrag { v16bf v; v8bf h[2]; };

static __device__ __forceinline__ v8f wmma_bf16(v16bf a, v16bf b, v8f c) {
  return __builtin_amdgcn_wmma_f32_16x16x32_bf16(false, a, false, b,
                                                 (short)0, c, false, false);
}

// Async global->LDS copy, 16B per lane. lds_off = LDS byte address (low 32
// bits of the generic shared pointer), g = per-lane global address.
static __device__ __forceinline__ void async_cp16(unsigned lds_off,
                                                  const void* g) {
  asm volatile("global_load_async_to_lds_b128 %0, %1, off"
               :: "v"(lds_off), "v"(g) : "memory");
}
static __device__ __forceinline__ void wait_async0() {
  asm volatile("s_wait_asynccnt 0x0" ::: "memory");
}

#define B_  2
#define T_  2048
#define NH_ 12
#define HD_ 128
#define H_  1536
#define EPS_ 1.1920928955078125e-07f

// ---------------------------------------------------------------------------
// f32 -> bf16 convert (vectorized x4)
// ---------------------------------------------------------------------------
__global__ void cvt_f32_bf16(const float* __restrict__ src,
                             __bf16* __restrict__ dst, int n) {
  int i = (blockIdx.x * blockDim.x + threadIdx.x) * 4;
  if (i + 3 < n) {
    float4 f = *(const float4*)(src + i);
    dst[i + 0] = (__bf16)f.x;
    dst[i + 1] = (__bf16)f.y;
    dst[i + 2] = (__bf16)f.z;
    dst[i + 3] = (__bf16)f.w;
  }
}

// ---------------------------------------------------------------------------
// C[M,N] = A[M,K] * W[N,K]^T   (both operands K-major; bf16 in, f32 out)
// 128x128 C-tile per 256-thread block (8 waves, each 32x64), K-step 32.
// Double-buffered LDS tiles filled by async global->LDS copies; one barrier
// per K-step. LDS row stride padded to 40 bf16 (80B): 16B aligned and
// bank-conflict-free for the 16-row strided ds_load_b128 fragment reads.
// ---------------------------------------------------------------------------
__global__ __launch_bounds__(256) void gemm_bf16(
    const __bf16* __restrict__ A, const __bf16* __restrict__ W,
    float* __restrict__ C, int M, int N, int K) {
  __shared__ __align__(16) __bf16 As[2][128 * 40];
  __shared__ __align__(16) __bf16 Bs[2][128 * 40];

  const int tid  = threadIdx.x;
  const int lane = tid & 31;
  const int hf   = lane >> 4;      // which 16-lane half
  const int ln   = lane & 15;
  const int wid  = tid >> 5;       // 0..7
  const int wm   = wid & 3;        // 4 wave-rows  (32 rows each)
  const int wn   = wid >> 2;       // 2 wave-cols  (64 cols each)
  const int m0   = blockIdx.y * 128;
  const int n0   = blockIdx.x * 128;

  // This thread's two staging chunks (row, kc) for each of A and B.
  const int row0 = tid >> 2,           kc0 = tid & 3;
  const int row1 = (tid + 256) >> 2,   kc1 = (tid + 256) & 3;

  auto stage = [&](int k0, int buf) {
    async_cp16((unsigned)(size_t)&As[buf][row0 * 40 + kc0 * 8],
               &A[(size_t)(m0 + row0) * K + k0 + kc0 * 8]);
    async_cp16((unsigned)(size_t)&Bs[buf][row0 * 40 + kc0 * 8],
               &W[(size_t)(n0 + row0) * K + k0 + kc0 * 8]);
    async_cp16((unsigned)(size_t)&As[buf][row1 * 40 + kc1 * 8],
               &A[(size_t)(m0 + row1) * K + k0 + kc1 * 8]);
    async_cp16((unsigned)(size_t)&Bs[buf][row1 * 40 + kc1 * 8],
               &W[(size_t)(n0 + row1) * K + k0 + kc1 * 8]);
  };

  v8f acc[2][4];
  for (int r = 0; r < 2; ++r)
    for (int c = 0; c < 4; ++c)
      for (int i = 0; i < 8; ++i) acc[r][c][i] = 0.f;

  stage(0, 0);
  int cur = 0;
  for (int k0 = 0; k0 < K; k0 += 32) {
    wait_async0();       // our async copies into buf[cur] have landed
    __syncthreads();     // everyone's copies have landed
    if (k0 + 32 < K) stage(k0 + 32, cur ^ 1);  // prefetch overlaps compute

    BFrag a[2], b[4];
    for (int r = 0; r < 2; ++r) {
      int rr = wm * 32 + r * 16 + ln;
      a[r].h[0] = *(const v8bf*)&As[cur][rr * 40 + 8 * hf];
      a[r].h[1] = *(const v8bf*)&As[cur][rr * 40 + 16 + 8 * hf];
    }
    for (int c = 0; c < 4; ++c) {
      int cc = wn * 64 + c * 16 + ln;
      b[c].h[0] = *(const v8bf*)&Bs[cur][cc * 40 + 8 * hf];
      b[c].h[1] = *(const v8bf*)&Bs[cur][cc * 40 + 16 + 8 * hf];
    }
    for (int r = 0; r < 2; ++r)
      for (int c = 0; c < 4; ++c)
        acc[r][c] = wmma_bf16(a[r].v, b[c].v, acc[r][c]);
    cur ^= 1;
  }

  // C layout: VGPR i -> row i (lanes 0-15) / row i+8 (lanes 16-31); col = ln.
  for (int r = 0; r < 2; ++r)
    for (int c = 0; c < 4; ++c)
      for (int i = 0; i < 8; ++i) {
        int m = m0 + wm * 32 + r * 16 + i + 8 * hf;
        int n = n0 + wn * 64 + c * 16 + ln;
        C[(size_t)m * N + n] = acc[r][c][i];
      }
}

// ---------------------------------------------------------------------------
// RoPE + RMSNorm on q,k; bf16 convert; V transposed to [b,h,d,T].
// One wave per (b,t,h) row of 128; lane handles 2 rotary pairs (4 elems).
// ---------------------------------------------------------------------------
__global__ __launch_bounds__(256) void rope_rms(
    const float* __restrict__ qf, const float* __restrict__ kf,
    const float* __restrict__ vf, const float* __restrict__ cosp,
    const float* __restrict__ sinp, __bf16* __restrict__ qb,
    __bf16* __restrict__ kb, __bf16* __restrict__ vT) {
  const int tid  = threadIdx.x;
  const int lane = tid & 31;
  const int row  = blockIdx.x * 8 + (tid >> 5);  // ((b*T+t)*NH + h)
  const int h    = row % NH_;
  const int t    = (row / NH_) % T_;
  const int b    = row / (NH_ * T_);
  const int p    = lane * 2;  // pairs p, p+1 in [0,64)

  const float* cp = cosp + (size_t)t * (HD_ / 2);
  const float* sp = sinp + (size_t)t * (HD_ / 2);
  const float ca = cp[p], cb = cp[p + 1], sa = sp[p], sb = sp[p + 1];

  // ---- Q ----
  {
    const float* x = qf + (size_t)row * HD_;
    float x1a = x[p], x1b = x[p + 1], x2a = x[p + 64], x2b = x[p + 65];
    float r1a = x1a * ca + x2a * sa, r2a = -x1a * sa + x2a * ca;
    float r1b = x1b * cb + x2b * sb, r2b = -x1b * sb + x2b * cb;
    float ss = r1a * r1a + r1b * r1b + r2a * r2a + r2b * r2b;
    for (int m = 1; m <= 16; m <<= 1) ss += __shfl_xor(ss, m, 32);
    float rn = rsqrtf(ss * (1.0f / HD_) + EPS_);
    __bf16* o = qb + (size_t)row * HD_;
    o[p] = (__bf16)(r1a * rn);       o[p + 1] = (__bf16)(r1b * rn);
    o[p + 64] = (__bf16)(r2a * rn);  o[p + 65] = (__bf16)(r2b * rn);
  }
  // ---- K ----
  {
    const float* x = kf + (size_t)row * HD_;
    float x1a = x[p], x1b = x[p + 1], x2a = x[p + 64], x2b = x[p + 65];
    float r1a = x1a * ca + x2a * sa, r2a = -x1a * sa + x2a * ca;
    float r1b = x1b * cb + x2b * sb, r2b = -x1b * sb + x2b * cb;
    float ss = r1a * r1a + r1b * r1b + r2a * r2a + r2b * r2b;
    for (int m = 1; m <= 16; m <<= 1) ss += __shfl_xor(ss, m, 32);
    float rn = rsqrtf(ss * (1.0f / HD_) + EPS_);
    __bf16* o = kb + (size_t)row * HD_;
    o[p] = (__bf16)(r1a * rn);       o[p + 1] = (__bf16)(r1b * rn);
    o[p + 64] = (__bf16)(r2a * rn);  o[p + 65] = (__bf16)(r2b * rn);
  }
  // ---- V: transpose to [b,h,d,T] for contiguous-key PV fragment loads ----
  {
    const float* x = vf + (size_t)row * HD_;
    __bf16* o = vT + ((size_t)(b * NH_ + h) * HD_) * T_ + t;
    o[(size_t)(p) * T_]      = (__bf16)x[p];
    o[(size_t)(p + 1) * T_]  = (__bf16)x[p + 1];
    o[(size_t)(p + 64) * T_] = (__bf16)x[p + 64];
    o[(size_t)(p + 65) * T_] = (__bf16)x[p + 65];
  }
}

// ---------------------------------------------------------------------------
// Flash attention: one wave per 16-query tile of one (b,h); online softmax
// over 32-key blocks inside the causal sliding window. P restaged via LDS
// (C-layout f32 -> A-layout bf16). 4 independent waves per block.
// ---------------------------------------------------------------------------
__global__ __launch_bounds__(128) void attn_fwd(
    const __bf16* __restrict__ qb, const __bf16* __restrict__ kb,
    const __bf16* __restrict__ vT, __bf16* __restrict__ yb,
    const int* __restrict__ winp) {
  __shared__ __align__(16) __bf16 Pl[4][16 * 40];  // per-wave P tile, padded

  const int tid  = threadIdx.x;
  const int lane = tid & 31;
  const int hf   = lane >> 4;
  const int ln   = lane & 15;
  const int w    = tid >> 5;
  const int gw   = blockIdx.x * 4 + w;     // global wave id
  const int qt   = gw & 127;               // T/16 = 128 query tiles
  const int hh   = (gw >> 7) % NH_;
  const int b    = gw / (128 * NH_);
  const int q0   = qt * 16;
  const int win  = *winp;
  const float scale = 0.08838834764831845f;  // 1/sqrt(128)

  // Q fragments: 16 rows x 128 (4 chunks of K=32)
  BFrag qfr[4];
  {
    const __bf16* p = qb + ((size_t)(b * T_ + q0 + ln) * H_) + hh * HD_;
    for (int c = 0; c < 4; ++c) {
      qfr[c].h[0] = *(const v8bf*)(p + c * 32 + 8 * hf);
      qfr[c].h[1] = *(const v8bf*)(p + c * 32 + 16 + 8 * hf);
    }
  }

  v8f o[8];
  float m_i[8], l_i[8];
  for (int dc = 0; dc < 8; ++dc)
    for (int i = 0; i < 8; ++i) o[dc][i] = 0.f;
  for (int i = 0; i < 8; ++i) { m_i[i] = -1e30f; l_i[i] = 0.f; }

  int kb0 = q0 - win;
  if (kb0 < 0) kb0 = 0;
  kb0 &= ~31;

  for (int kk = kb0; kk <= q0 + 15; kk += 32) {
    // ---- S = Q * K^T  (two 16-key n-tiles) ----
    v8f s0, s1;
    for (int i = 0; i < 8; ++i) { s0[i] = 0.f; s1[i] = 0.f; }
    for (int c = 0; c < 4; ++c) {
      BFrag kf0, kf1;
      const __bf16* kp =
          kb + ((size_t)(b * T_ + kk + ln) * H_) + hh * HD_ + c * 32;
      kf0.h[0] = *(const v8bf*)(kp + 8 * hf);
      kf0.h[1] = *(const v8bf*)(kp + 16 + 8 * hf);
      const __bf16* kp1 = kp + (size_t)16 * H_;
      kf1.h[0] = *(const v8bf*)(kp1 + 8 * hf);
      kf1.h[1] = *(const v8bf*)(kp1 + 16 + 8 * hf);
      s0 = wmma_bf16(qfr[c].v, kf0.v, s0);
      s1 = wmma_bf16(qfr[c].v, kf1.v, s1);
    }

    // ---- mask + online softmax (stats per row i+8*hf, dup across 16 lanes)
    for (int i = 0; i < 8; ++i) {
      int row = q0 + i + 8 * hf;
      int c0 = kk + ln, c1 = kk + 16 + ln;
      bool ok0 = (c0 <= row) && (row - c0 <= win);
      bool ok1 = (c1 <= row) && (row - c1 <= win);
      float v0 = ok0 ? s0[i] * scale : -1e30f;
      float v1 = ok1 ? s1[i] * scale : -1e30f;
      float mx = fmaxf(v0, v1);
      mx = fmaxf(mx, __shfl_xor(mx, 1, 32));
      mx = fmaxf(mx, __shfl_xor(mx, 2, 32));
      mx = fmaxf(mx, __shfl_xor(mx, 4, 32));
      mx = fmaxf(mx, __shfl_xor(mx, 8, 32));
      float mn = fmaxf(m_i[i], mx);
      float alpha = __expf(m_i[i] - mn);
      m_i[i] = mn;
      float p0 = __expf(v0 - mn), p1 = __expf(v1 - mn);
      float rs = p0 + p1;
      rs += __shfl_xor(rs, 1, 32);
      rs += __shfl_xor(rs, 2, 32);
      rs += __shfl_xor(rs, 4, 32);
      rs += __shfl_xor(rs, 8, 32);
      l_i[i] = l_i[i] * alpha + rs;
      for (int dc = 0; dc < 8; ++dc) o[dc][i] *= alpha;
      // stash P (C-layout -> LDS, padded stride 40 bf16)
      Pl[w][(i + 8 * hf) * 40 + ln] = (__bf16)p0;
      Pl[w][(i + 8 * hf) * 40 + 16 + ln] = (__bf16)p1;
    }

    // ---- O += P * V  (P A-frag from LDS; V B-frag contiguous along T) ----
    BFrag pf;
    pf.h[0] = *(const v8bf*)&Pl[w][ln * 40 + 8 * hf];
    pf.h[1] = *(const v8bf*)&Pl[w][ln * 40 + 16 + 8 * hf];
    const __bf16* vb = vT + ((size_t)(b * NH_ + hh) * HD_) * T_ + kk;
    for (int dc = 0; dc < 8; ++dc) {
      BFrag vf;
      const __bf16* vp = vb + (size_t)(dc * 16 + ln) * T_;
      vf.h[0] = *(const v8bf*)(vp + 8 * hf);
      vf.h[1] = *(const v8bf*)(vp + 16 + 8 * hf);
      o[dc] = wmma_bf16(pf.v, vf.v, o[dc]);
    }
  }

  // ---- normalize and store bf16 [tok, H] ----
  for (int i = 0; i < 8; ++i) {
    float inv = 1.0f / l_i[i];
    for (int dc = 0; dc < 8; ++dc) o[dc][i] *= inv;
  }
  for (int dc = 0; dc < 8; ++dc)
    for (int i = 0; i < 8; ++i) {
      int row = q0 + i + 8 * hf;
      yb[(size_t)(b * T_ + row) * H_ + hh * HD_ + dc * 16 + ln] =
          (__bf16)o[dc][i];
    }
}

// ---------------------------------------------------------------------------
// Host orchestration
// ---------------------------------------------------------------------------
extern "C" void kernel_launch(void* const* d_in, const int* in_sizes, int n_in,
                              void* d_out, int out_size, void* d_ws,
                              size_t ws_size, hipStream_t stream) {
  const float* x    = (const float*)d_in[0];
  const float* cosp = (const float*)d_in[1];
  const float* sinp = (const float*)d_in[2];
  const float* Wq   = (const float*)d_in[3];
  const float* Wk   = (const float*)d_in[4];
  const float* Wv   = (const float*)d_in[5];
  const float* Wp   = (const float*)d_in[6];
  const int*   winp = (const int*)d_in[7];
  float* out = (float*)d_out;

  const size_t Mtok = (size_t)B_ * T_;          // 4096
  const size_t szB  = Mtok * H_ * 2;            // bf16 activation
  const size_t szW  = (size_t)H_ * H_ * 2;      // bf16 weight
  const size_t szF  = Mtok * H_ * 4;            // f32 activation

  char* ws = (char*)d_ws;
  size_t off = 0;
  auto carve = [&](size_t bytes) {
    void* p = ws + off;
    off += (bytes + 255) & ~(size_t)255;
    return p;
  };
  __bf16* xb  = (__bf16*)carve(szB);
  __bf16* wqb = (__bf16*)carve(szW);
  __bf16* wkb = (__bf16*)carve(szW);
  __bf16* wvb = (__bf16*)carve(szW);
  __bf16* wpb = (__bf16*)carve(szW);
  float*  qf  = (float*)carve(szF);
  float*  kf  = (float*)carve(szF);
  float*  vf  = (float*)carve(szF);
  __bf16* qbb = (__bf16*)carve(szB);
  __bf16* kbb = (__bf16*)carve(szB);
  __bf16* vT  = (__bf16*)carve(szB);
  __bf16* yb  = (__bf16*)qf;  // qf dead after rope_rms; reuse for attn output

  // 1) f32 -> bf16 converts
  {
    int n = (int)(Mtok * H_);
    cvt_f32_bf16<<<n / 1024, 256, 0, stream>>>(x, xb, n);
    int nw = H_ * H_;
    cvt_f32_bf16<<<nw / 1024, 256, 0, stream>>>(Wq, wqb, nw);
    cvt_f32_bf16<<<nw / 1024, 256, 0, stream>>>(Wk, wkb, nw);
    cvt_f32_bf16<<<nw / 1024, 256, 0, stream>>>(Wv, wvb, nw);
    cvt_f32_bf16<<<nw / 1024, 256, 0, stream>>>(Wp, wpb, nw);
  }

  // 2) QKV projections: [4096,1536] = xb * W^T
  dim3 gg(H_ / 128, (int)Mtok / 128);  // (12, 32)
  gemm_bf16<<<gg, 256, 0, stream>>>(xb, wqb, qf, (int)Mtok, H_, H_);
  gemm_bf16<<<gg, 256, 0, stream>>>(xb, wkb, kf, (int)Mtok, H_, H_);
  gemm_bf16<<<gg, 256, 0, stream>>>(xb, wvb, vf, (int)Mtok, H_, H_);

  // 3) RoPE + RMSNorm (+ V transpose): B*T*NH rows, 8 waves/block
  rope_rms<<<(B_ * T_ * NH_) / 8, 256, 0, stream>>>(qf, kf, vf, cosp, sinp,
                                                    qbb, kbb, vT);

  // 4) Flash attention: B*NH*(T/16) wave-tiles, 4 waves/block
  attn_fwd<<<(B_ * NH_ * (T_ / 16)) / 4, 128, 0, stream>>>(qbb, kbb, vT, yb,
                                                           winp);

  // 5) Output projection -> f32 d_out
  gemm_bf16<<<gg, 256, 0, stream>>>(yb, wpb, out, (int)Mtok, H_, H_);
}
